// MuSc_59983513256517
// MI455X (gfx1250) — compile-verified, matrix operand order checked by reading, not run on hardware
//
#include <hip/hip_runtime.h>
#include <hip/hip_bf16.h>

typedef __attribute__((ext_vector_type(16))) _Float16 v16h;
typedef __attribute__((ext_vector_type(8)))  float    v8f;

#define LNUM 4
#define BNUM 8
#define PNUM 784
#define DNUM 1024
#define PHW  28
#define NPAIR 56          // 8*7 ordered pairs, c != b
#define PTILES 49         // 784 / 16
#define MTILES 4          // 64 rows per block
#define NTILES 2          // 32 cols per wave iteration
#define CGRP   25         // ceil(49/2) column groups
#define RBLK   13         // ceil(49/4) row blocks per pair

// ---------------------------------------------------------------------------
// 1) LayerNorm over D for every (l,b,p) token. fp32 in -> fp16 out.
// ---------------------------------------------------------------------------
__global__ __launch_bounds__(256) void ln_kernel(const float* __restrict__ feat,
                                                 _Float16* __restrict__ out) {
    const int row = blockIdx.x;                    // (l*B + b)*P + p
    const float* x = feat + (size_t)row * DNUM;
    _Float16*    y = out  + (size_t)row * DNUM;
    const int t = threadIdx.x;

    float v[4];
    float s = 0.f, s2 = 0.f;
#pragma unroll
    for (int i = 0; i < 4; ++i) {
        v[i] = x[t + 256 * i];
        s  += v[i];
        s2 += v[i] * v[i];
    }
    for (int off = 16; off; off >>= 1) {
        s  += __shfl_down(s,  off, 32);
        s2 += __shfl_down(s2, off, 32);
    }
    __shared__ float red[8][2];
    const int wave = t >> 5, lane = t & 31;
    if (lane == 0) { red[wave][0] = s; red[wave][1] = s2; }
    __syncthreads();
    float ts = 0.f, ts2 = 0.f;
#pragma unroll
    for (int w = 0; w < 8; ++w) { ts += red[w][0]; ts2 += red[w][1]; }

    const float mu  = ts * (1.0f / DNUM);
    const float var = ts2 * (1.0f / DNUM) - mu * mu;
    const float rs  = rsqrtf(var + 1e-5f);
#pragma unroll
    for (int i = 0; i < 4; ++i)
        y[t + 256 * i] = (_Float16)((v[i] - mu) * rs);
}

// ---------------------------------------------------------------------------
// 2) r x r box-sum over patch grid (zero pad) + L2 normalize -> fp16 fhat.
//    The 1/r^2 factor cancels in the normalization.
// ---------------------------------------------------------------------------
__global__ __launch_bounds__(256) void norm_kernel(const _Float16* __restrict__ ln,
                                                   _Float16* __restrict__ fhat,
                                                   int r) {
    const int bp = blockIdx.x;                     // b*P + p
    const int b  = bp / PNUM, p = bp % PNUM;
    const int py = p / PHW,  px = p % PHW;
    const int t  = threadIdx.x;
    const int h  = r >> 1;

    float v[4] = {0.f, 0.f, 0.f, 0.f};
    for (int dy = -h; dy <= h; ++dy) {
        const int qy = py + dy;
        if ((unsigned)qy >= (unsigned)PHW) continue;
        for (int dx = -h; dx <= h; ++dx) {
            const int qx = px + dx;
            if ((unsigned)qx >= (unsigned)PHW) continue;
            const _Float16* src = ln + ((size_t)(b * PNUM + qy * PHW + qx)) * DNUM;
#pragma unroll
            for (int i = 0; i < 4; ++i) v[i] += (float)src[t + 256 * i];
        }
    }
    float s2 = 0.f;
#pragma unroll
    for (int i = 0; i < 4; ++i) s2 += v[i] * v[i];
    for (int off = 16; off; off >>= 1) s2 += __shfl_down(s2, off, 32);
    __shared__ float red[8];
    const int wave = t >> 5, lane = t & 31;
    if (lane == 0) red[wave] = s2;
    __syncthreads();
    float ts2 = 0.f;
#pragma unroll
    for (int w = 0; w < 8; ++w) ts2 += red[w];

    const float rs = rsqrtf(fmaxf(ts2, 1e-24f));
    _Float16* dst = fhat + (size_t)bp * DNUM;
#pragma unroll
    for (int i = 0; i < 4; ++i) dst[t + 256 * i] = (_Float16)(v[i] * rs);
}

// ---------------------------------------------------------------------------
// 3) Row-max-fused gram GEMM, 64x32 register-blocked:
//    block = 64 query rows (4 M-tiles in LDS) x all 784 cols of image c.
//    Each wave computes a 4x2 grid of WMMA tiles per K-step:
//      8 wmma from 8 ds_load_b128 + 4 global_load_b128
//    -> 1 ds_b128 and 0.5 global_b128 per WMMA.
// ---------------------------------------------------------------------------
__global__ __launch_bounds__(256) void grammax_kernel(const _Float16* __restrict__ F,
                                                      float* __restrict__ maxsim) {
    const int blk  = blockIdx.x;
    const int pair = blk / RBLK, rb = blk % RBLK;
    const int b  = pair / 7, tt = pair % 7;
    const int c  = tt + (tt >= b ? 1 : 0);

    const int lane = threadIdx.x & 31;
    const int wave = threadIdx.x >> 5;             // 0..7

    __shared__ __align__(16) _Float16 As[64][1032];   // 129 KB, padded stride
    __shared__ float part[8][64];

    // cooperative load of 64 A rows (row index clamped at 783; duplicated
    // rows are harmless for a row-max and never written out)
    for (int i = threadIdx.x; i < 64 * 128; i += 256) {
        const int row = i >> 7, kv = i & 127;
        int grow = rb * 64 + row;
        if (grow > PNUM - 1) grow = PNUM - 1;
        *(uint4*)(&As[row][kv * 8]) =
            *(const uint4*)(F + ((size_t)(b * PNUM + grow)) * DNUM + kv * 8);
    }
    __syncthreads();

    const int arow = lane & 15;        // row within an M-tile held by this lane
    const int kh   = lane >> 4;        // K half selector per ISA layout

    float rmax[MTILES][8];
#pragma unroll
    for (int mt = 0; mt < MTILES; ++mt)
#pragma unroll
        for (int i = 0; i < 8; ++i) rmax[mt][i] = -2.0f;

    for (int cg = wave; cg < CGRP; cg += 8) {
        // two adjacent column tiles; odd trailing tile clamps to column 783
        // (duplicate of a valid column: no effect on the row max)
        const int q0 = (cg * 2 + 0) * 16 + (lane & 15);
        int       q1 = (cg * 2 + 1) * 16 + (lane & 15);
        if (q1 > PNUM - 1) q1 = PNUM - 1;
        const _Float16* B0 = F + ((size_t)(c * PNUM + q0)) * DNUM;
        const _Float16* B1 = F + ((size_t)(c * PNUM + q1)) * DNUM;

        v8f acc[MTILES][NTILES] = {};
#pragma unroll 2
        for (int kc = 0; kc < 32; ++kc) {
            const int k0 = kc * 32;
            union { uint4 u[2]; v16h h; } bf0, bf1;
            bf0.u[0] = *(const uint4*)(B0 + k0 + kh * 8);
            bf0.u[1] = *(const uint4*)(B0 + k0 + 16 + kh * 8);
            bf1.u[0] = *(const uint4*)(B1 + k0 + kh * 8);
            bf1.u[1] = *(const uint4*)(B1 + k0 + 16 + kh * 8);
#pragma unroll
            for (int mt = 0; mt < MTILES; ++mt) {
                union { uint4 u[2]; v16h h; } af;
                af.u[0] = *(const uint4*)(&As[mt * 16 + arow][k0 + kh * 8]);
                af.u[1] = *(const uint4*)(&As[mt * 16 + arow][k0 + 16 + kh * 8]);
                acc[mt][0] = __builtin_amdgcn_wmma_f32_16x16x32_f16(
                    false, af.h, false, bf0.h, (short)0, acc[mt][0], false, false);
                acc[mt][1] = __builtin_amdgcn_wmma_f32_16x16x32_f16(
                    false, af.h, false, bf1.h, (short)0, acc[mt][1], false, false);
            }
        }
#pragma unroll
        for (int mt = 0; mt < MTILES; ++mt)
#pragma unroll
            for (int i = 0; i < 8; ++i)
                rmax[mt][i] = fmaxf(rmax[mt][i],
                                    fmaxf(acc[mt][0][i], acc[mt][1][i]));
    }

    // butterfly max across each 16-lane half (N dimension of the tile)
#pragma unroll
    for (int off = 8; off; off >>= 1)
#pragma unroll
        for (int mt = 0; mt < MTILES; ++mt)
#pragma unroll
            for (int i = 0; i < 8; ++i)
                rmax[mt][i] = fmaxf(rmax[mt][i], __shfl_xor(rmax[mt][i], off, 32));

    // C layout: VGPR i = row i (lanes 0-15) / row i+8 (lanes 16-31)
    if ((lane & 15) == 0) {
        const int base = (lane >> 4) * 8;
#pragma unroll
        for (int mt = 0; mt < MTILES; ++mt)
#pragma unroll
            for (int i = 0; i < 8; ++i)
                part[wave][mt * 16 + base + i] = rmax[mt][i];
    }
    __syncthreads();
    if (threadIdx.x < 64) {
        float m = part[0][threadIdx.x];
#pragma unroll
        for (int w = 1; w < 8; ++w) m = fmaxf(m, part[w][threadIdx.x]);
        const int grow = rb * 64 + threadIdx.x;
        if (grow < PNUM)
            maxsim[((size_t)(b * BNUM + c)) * PNUM + grow] = m;
    }
}

// ---------------------------------------------------------------------------
// 4) dist = sqrt(2 - 2*maxsim); mean of the 2 smallest refs; accumulate.
// ---------------------------------------------------------------------------
__global__ __launch_bounds__(256) void topk_kernel(const float* __restrict__ maxsim,
                                                   float* __restrict__ scores,
                                                   float w, int first) {
    const int i = blockIdx.x * blockDim.x + threadIdx.x;
    if (i >= BNUM * PNUM) return;
    const int b = i / PNUM, p = i % PNUM;
    float d1 = 1e30f, d2 = 1e30f;
#pragma unroll
    for (int cc = 0; cc < BNUM; ++cc) {
        if (cc == b) continue;
        const float m = maxsim[((size_t)(b * BNUM + cc)) * PNUM + p];
        const float d = sqrtf(fmaxf(2.0f - 2.0f * m, 1e-12f));
        if (d < d1) { d2 = d1; d1 = d; }
        else if (d < d2) d2 = d;
    }
    const float s = 0.5f * (d1 + d2) * w;
    if (first) scores[i] = s;
    else       scores[i] += s;
}

// ---------------------------------------------------------------------------
// 5) per-image max over 784 patches.
// ---------------------------------------------------------------------------
__global__ __launch_bounds__(256) void imgmax_kernel(const float* __restrict__ scores,
                                                     float* __restrict__ out) {
    const int b = blockIdx.x, t = threadIdx.x;
    float m = -1e30f;
    for (int p = t; p < PNUM; p += 256) m = fmaxf(m, scores[b * PNUM + p]);
    for (int off = 16; off; off >>= 1) m = fmaxf(m, __shfl_down(m, off, 32));
    __shared__ float red[8];
    if ((t & 31) == 0) red[t >> 5] = m;
    __syncthreads();
    if (t == 0) {
        float r = red[0];
#pragma unroll
        for (int w = 1; w < 8; ++w) r = fmaxf(r, red[w]);
        out[b] = r;
    }
}

// ---------------------------------------------------------------------------
// 6) bilinear upsample 28x28 -> 392x392, align_corners=True.
// ---------------------------------------------------------------------------
__global__ __launch_bounds__(256) void upsample_kernel(const float* __restrict__ scores,
                                                       float* __restrict__ out) {
    const int HW = 392 * 392;
    const int i = blockIdx.x * blockDim.x + threadIdx.x;
    if (i >= BNUM * HW) return;
    const int b = i / HW, rem = i % HW;
    const int y = rem / 392, x = rem % 392;
    const float sc = 27.0f / 391.0f;
    const float fy = y * sc, fx = x * sc;
    int y0 = (int)fy, x0 = (int)fx;
    int y1 = y0 + 1 > 27 ? 27 : y0 + 1;
    int x1 = x0 + 1 > 27 ? 27 : x0 + 1;
    const float wy = fy - (float)y0, wx = fx - (float)x0;
    const float* s = scores + b * PNUM;
    const float g00 = s[y0 * PHW + x0], g01 = s[y0 * PHW + x1];
    const float g10 = s[y1 * PHW + x0], g11 = s[y1 * PHW + x1];
    out[i] = g00 * (1.f - wy) * (1.f - wx) + g01 * (1.f - wy) * wx
           + g10 * wy * (1.f - wx) + g11 * wy * wx;
}

// ---------------------------------------------------------------------------
extern "C" void kernel_launch(void* const* d_in, const int* in_sizes, int n_in,
                              void* d_out, int out_size, void* d_ws, size_t ws_size,
                              hipStream_t stream) {
    (void)in_sizes; (void)n_in; (void)out_size; (void)ws_size;
    const float* feat = (const float*)d_in[0];      // [L,B,P,D] fp32
    float* out = (float*)d_out;                     // [8] ++ [8,392,392]

    char* ws = (char*)d_ws;
    size_t off = 0;
    _Float16* ln = (_Float16*)(ws + off);
    off += (size_t)LNUM * BNUM * PNUM * DNUM * sizeof(_Float16);   // 51.4 MB
    off = (off + 255) & ~(size_t)255;
    _Float16* fhat = (_Float16*)(ws + off);
    off += (size_t)BNUM * PNUM * DNUM * sizeof(_Float16);          // 12.85 MB
    off = (off + 255) & ~(size_t)255;
    float* maxsim = (float*)(ws + off);
    off += (size_t)BNUM * BNUM * PNUM * sizeof(float);             // 200 KB
    off = (off + 255) & ~(size_t)255;
    float* scores = (float*)(ws + off);                            // 25 KB

    ln_kernel<<<LNUM * BNUM * PNUM, 256, 0, stream>>>(feat, ln);

    const int rlist[3] = {1, 3, 5};
    int combo = 0;
    for (int ri = 0; ri < 3; ++ri) {
        for (int l = 0; l < LNUM; ++l, ++combo) {
            norm_kernel<<<BNUM * PNUM, 256, 0, stream>>>(
                ln + (size_t)l * BNUM * PNUM * DNUM, fhat, rlist[ri]);
            grammax_kernel<<<NPAIR * RBLK, 256, 0, stream>>>(fhat, maxsim);
            topk_kernel<<<(BNUM * PNUM + 255) / 256, 256, 0, stream>>>(
                maxsim, scores, 1.0f / 12.0f, combo == 0 ? 1 : 0);
        }
    }

    imgmax_kernel<<<BNUM, 256, 0, stream>>>(scores, out);
    upsample_kernel<<<(BNUM * 392 * 392 + 255) / 256, 256, 0, stream>>>(scores, out + 8);
}
